// SCOOP_77464030150827
// MI455X (gfx1250) — compile-verified, matrix operand order checked by reading, not run on hardware
//
#include <hip/hip_runtime.h>
#include <hip/hip_bf16.h>

// MI455X / gfx1250 SCOOP forward pass.
// All GEMMs run on v_wmma_f32_16x16x32_f16 with fully vectorized fragment
// loads (two contiguous 16B runs per lane-half). The 2x4096x4096 Sinkhorn
// kernel matrix (128MB f32) is L2-resident (192MB L2) across 20 matvec passes.

typedef __attribute__((ext_vector_type(16))) _Float16 v16h;
typedef __attribute__((ext_vector_type(8)))  _Float16 h8;
typedef __attribute__((ext_vector_type(8)))  float    v8f;
typedef __attribute__((ext_vector_type(4)))  float    f4;

#define BB 2
#define NN 4096
#define MM 4096
#define KNB 32          // KNN graph k
#define KREC 32         // top-k reconstruction
#define NB_ITER 10
#define ROWS_PER_B (NN*KNB)        // 131072
#define ROWS_TOTAL (BB*ROWS_PER_B) // 262144

// ---------------------------------------------------------------- KNN graph
// LDS-tiled: block streams 256-point chunks of the cloud through shared mem.
__global__ void knn_kernel(const float* __restrict__ pc, int* __restrict__ idxOut) {
  __shared__ float sp[256 * 3];
  int t = blockIdx.x * blockDim.x + threadIdx.x;   // BB*NN threads, 256/block
  int b = t / NN, n = t % NN;                      // block never straddles b
  const float* base = pc + (size_t)b * NN * 3;
  float px = base[n*3+0], py = base[n*3+1], pz = base[n*3+2];
  float dist[KNB]; int ind[KNB];
  for (int i = 0; i < KNB; i++) { dist[i] = 3.4e38f; ind[i] = 0; }
  for (int j0 = 0; j0 < NN; j0 += 256) {
    __syncthreads();
    for (int u = threadIdx.x; u < 256 * 3; u += 256) sp[u] = base[j0*3 + u];
    __syncthreads();
    for (int jj = 0; jj < 256; jj++) {
      float dx = sp[jj*3+0]-px, dy = sp[jj*3+1]-py, dz = sp[jj*3+2]-pz;
      float d = dx*dx + dy*dy + dz*dz;
      if (d < dist[KNB-1]) {
        int pos = KNB-1;
        while (pos > 0 && dist[pos-1] > d) {
          dist[pos] = dist[pos-1]; ind[pos] = ind[pos-1]; pos--;
        }
        dist[pos] = d; ind[pos] = j0 + jj;
      }
    }
  }
  int* o = idxOut + (size_t)t * KNB;
  for (int i = 0; i < KNB; i++) o[i] = ind[i];
}

// ----------------------- gather + edge concat, padded to CinPad (f32 rows)
__global__ void gather_concat(const float* __restrict__ sig, const int* __restrict__ idx,
                              const float* __restrict__ pc, float* __restrict__ Ag,
                              int sigC, int CinPad) {
  size_t total = (size_t)ROWS_TOTAL * CinPad;
  size_t t = (size_t)blockIdx.x * blockDim.x + threadIdx.x;
  if (t >= total) return;
  int c = (int)(t % CinPad);
  size_t row = t / CinPad;                 // row == (b*NN+n)*KNB + k
  int b = (int)(row / ROWS_PER_B);
  int rem = (int)(row % ROWS_PER_B);
  int n = rem / KNB;
  int g = idx[row];
  float v = 0.f;
  if (c < sigC) v = sig[(size_t)(b*NN + g) * sigC + c];
  else if (c < sigC + 3) {
    int d = c - sigC;
    v = pc[(size_t)(b*NN + g)*3 + d] - pc[(size_t)(b*NN + n)*3 + d];
  }
  Ag[t] = v;
}

// ---------------- weight pack: W[Cin][Cout] f32 -> Wt[Cout][CinPad] f16
__global__ void wt_pack(const float* __restrict__ W, _Float16* __restrict__ Wt,
                        int Cin, int CinPad, int Cout) {
  int t = blockIdx.x * blockDim.x + threadIdx.x;
  if (t >= Cout * CinPad) return;
  int col = t / CinPad, kd = t % CinPad;
  Wt[t] = (_Float16)(kd < Cin ? W[(size_t)kd * Cout + col] : 0.0f);
}

// ----------------------------------- uniform WMMA GEMM: Y = A * Wt^T + bias
// A: rows x CinPad f32 (CinPad % 32 == 0), Wt: [Cout][CinPad] f16.
// Lane-half h fragment = two contiguous 8-elem runs at k0+8h and k0+16+8h.
__global__ void gemm_wmma(const float* __restrict__ A, const _Float16* __restrict__ Wt,
                          const float* __restrict__ bias, float* __restrict__ Y,
                          int CinPad, int Cout) {
  const int RT = ROWS_TOTAL / 16;
  const int CT = Cout / 16;                 // Cout in {32,64,128}
  int wid = (blockIdx.x * blockDim.x + threadIdx.x) >> 5;
  if (wid >= RT * CT) return;
  int tr = wid / CT, tc = wid % CT;
  int lane = threadIdx.x & 31;
  int m16 = lane & 15, h = lane >> 4;
  int col = tc * 16 + m16;

  const float*    arow = A  + (size_t)(tr * 16 + m16) * CinPad;
  const _Float16* bcol = Wt + (size_t)col * CinPad;

  v8f acc = {};
  for (int k0 = 0; k0 < CinPad; k0 += 32) {
    const float* ap0 = arow + k0 + (h << 3);
    const float* ap1 = ap0 + 16;
    f4 x0 = *(const f4*)(ap0);
    f4 x1 = *(const f4*)(ap0 + 4);
    f4 x2 = *(const f4*)(ap1);
    f4 x3 = *(const f4*)(ap1 + 4);
    v16h af;
    #pragma unroll
    for (int e = 0; e < 4; e++) {
      af[e]      = (_Float16)x0[e];
      af[4 + e]  = (_Float16)x1[e];
      af[8 + e]  = (_Float16)x2[e];
      af[12 + e] = (_Float16)x3[e];
    }
    h8 b0 = *(const h8*)(bcol + k0 + (h << 3));
    h8 b1 = *(const h8*)(bcol + k0 + 16 + (h << 3));
    v16h bf;
    #pragma unroll
    for (int e = 0; e < 8; e++) { bf[e] = b0[e]; bf[8 + e] = b1[e]; }
    acc = __builtin_amdgcn_wmma_f32_16x16x32_f16(false, af, false, bf,
                                                 (short)0, acc, false, false);
  }
  float bs = bias[col];
  #pragma unroll
  for (int r = 0; r < 8; r++) {
    int orow = tr * 16 + r + (h << 3);
    Y[(size_t)orow * Cout + col] = acc[r] + bs;
  }
}

// ------------------------------------------------ instance-norm statistics
__global__ void zero_stats(float* __restrict__ stats, int n) {
  int t = blockIdx.x * blockDim.x + threadIdx.x;
  if (t < n) stats[t] = 0.f;
}

__global__ void colstats(const float* __restrict__ Y, float* __restrict__ stats, int C) {
  int blocksPerB = ROWS_PER_B / 1024;
  int b   = blockIdx.x / blocksPerB;
  int blk = blockIdx.x % blocksPerB;
  int t = threadIdx.x;
  int c = t % C;
  int rsub = t / C;
  int step = 256 / C;                 // C in {32,64,128} divides 256
  size_t base = (size_t)b * ROWS_PER_B + (size_t)blk * 1024;
  float s = 0.f, s2 = 0.f;
  for (int i = rsub; i < 1024; i += step) {
    float v = Y[(base + i) * C + c];
    s += v; s2 += v * v;
  }
  atomicAdd(&stats[(b * C + c) * 2 + 0], s);
  atomicAdd(&stats[(b * C + c) * 2 + 1], s2);
}

__global__ void norm_act(float* __restrict__ Y, const float* __restrict__ stats,
                         const float* __restrict__ g, const float* __restrict__ be, int C) {
  size_t total = (size_t)ROWS_TOTAL * C;
  size_t t = (size_t)blockIdx.x * blockDim.x + threadIdx.x;
  if (t >= total) return;
  int c = (int)(t % C);
  int b = (int)(t / ((size_t)ROWS_PER_B * C));
  const float cnt = (float)ROWS_PER_B;
  float su = stats[(b * C + c) * 2 + 0];
  float sq = stats[(b * C + c) * 2 + 1];
  float mean = su / cnt;
  float var  = sq / cnt - mean * mean;
  float v = (Y[t] - mean) * rsqrtf(var + 1e-5f) * g[c] + be[c];
  Y[t] = (v > 0.f) ? v : 0.1f * v;     // leaky relu 0.1
}

// ------------------------------------------------------------- max over K
__global__ void kmax_kernel(const float* __restrict__ Y, float* __restrict__ sig, int C) {
  size_t total = (size_t)BB * NN * C;
  size_t t = (size_t)blockIdx.x * blockDim.x + threadIdx.x;
  if (t >= total) return;
  int c = (int)(t % C);
  size_t bn = t / C;
  float mx = -3.4e38f;
  for (int k = 0; k < KNB; k++)
    mx = fmaxf(mx, Y[(bn * KNB + k) * C + c]);
  sig[t] = mx;
}

// -------------------------------------------- L2 normalize + f16 packing
__global__ void l2norm_pack(const float* __restrict__ sig,
                            float* __restrict__ fn, _Float16* __restrict__ fh) {
  __shared__ float red[128];
  int row = blockIdx.x;               // 0 .. BB*NN-1
  int t = threadIdx.x;                // 128 threads = 128 channels
  float v = sig[(size_t)row * 128 + t];
  red[t] = v * v;
  __syncthreads();
  for (int s = 64; s > 0; s >>= 1) {
    if (t < s) red[t] += red[t + s];
    __syncthreads();
  }
  float rn = rsqrtf(red[0] + 1e-8f);
  float nv = v * rn;
  fn[(size_t)row * 128 + t] = nv;
  fh[(size_t)row * 128 + t] = (_Float16)nv;
}

// --------------------- similarity GEMM (WMMA, K=128) fused into K-matrix
__global__ void sim_k_wmma(const _Float16* __restrict__ f1h,
                           const _Float16* __restrict__ f2h,
                           const float* __restrict__ pc1,
                           const float* __restrict__ pc2,
                           const float* __restrict__ epsP,
                           float* __restrict__ Kmat) {
  const int RT = NN / 16, CT = MM / 16;
  int wid = (blockIdx.x * blockDim.x + threadIdx.x) >> 5;
  if (wid >= BB * RT * CT) return;
  int b   = wid / (RT * CT);
  int rem = wid % (RT * CT);
  int tr = rem / CT, tc = rem % CT;
  int lane = threadIdx.x & 31;
  int m16 = lane & 15, h = lane >> 4;

  const _Float16* aPtr = f1h + (size_t)(b * NN + tr * 16 + m16) * 128;
  const _Float16* bPtr = f2h + (size_t)(b * MM + tc * 16 + m16) * 128;

  v8f acc = {};
  #pragma unroll
  for (int k0 = 0; k0 < 128; k0 += 32) {
    h8 a0 = *(const h8*)(aPtr + k0 + (h << 3));
    h8 a1 = *(const h8*)(aPtr + k0 + 16 + (h << 3));
    h8 b0 = *(const h8*)(bPtr + k0 + (h << 3));
    h8 b1 = *(const h8*)(bPtr + k0 + 16 + (h << 3));
    v16h af, bf;
    #pragma unroll
    for (int e = 0; e < 8; e++) {
      af[e] = a0[e]; af[8 + e] = a1[e];
      bf[e] = b0[e]; bf[8 + e] = b1[e];
    }
    acc = __builtin_amdgcn_wmma_f32_16x16x32_f16(false, af, false, bf,
                                                 (short)0, acc, false, false);
  }
  float eps = expf(epsP[0]) + 0.03f;
  int col = tc * 16 + m16;
  const float* q2 = pc2 + (size_t)(b * MM + col) * 3;
  float qx = q2[0], qy = q2[1], qz = q2[2];
  #pragma unroll
  for (int r = 0; r < 8; r++) {
    int row = tr * 16 + r + (h << 3);
    const float* q1 = pc1 + (size_t)(b * NN + row) * 3;
    float dx = q1[0]-qx, dy = q1[1]-qy, dz = q1[2]-qz;
    float d = dx*dx + dy*dy + dz*dz;
    float kv = (d < 100.0f) ? expf((acc[r] - 1.0f) / eps) : 0.0f;
    Kmat[(size_t)(b * NN + row) * MM + col] = kv;
  }
}

// ----------------------------------------------------------- Sinkhorn steps
__global__ void init_ab(float* __restrict__ avec, float* __restrict__ bvec) {
  int t = blockIdx.x * blockDim.x + threadIdx.x;
  if (t < BB * NN) avec[t] = 1.0f / NN;
  if (t < BB * MM) bvec[t] = 1.0f;
}

// b = (prob2 / (K^T a + 1e-8))^power — thread per column, a staged in LDS.
__global__ void sinkhorn_col(const float* __restrict__ Kmat, const float* __restrict__ avec,
                             float* __restrict__ bvec,
                             const float* __restrict__ epsP, const float* __restrict__ gamP) {
  __shared__ float sa[256];
  int t = blockIdx.x * blockDim.x + threadIdx.x;   // BB*MM threads
  int b = t / MM, m = t % MM;
  const float* kb = Kmat + (size_t)b * NN * MM + m;
  const float* ab = avec + b * NN;
  float acc = 0.f;
  for (int n0 = 0; n0 < NN; n0 += 256) {
    __syncthreads();
    sa[threadIdx.x] = ab[n0 + threadIdx.x];
    __syncthreads();
    for (int j = 0; j < 256; j++) {
      int n = n0 + j;
      if (n + 8 < NN) __builtin_prefetch(kb + (size_t)(n + 8) * MM, 0, 1);
      acc += kb[(size_t)n * MM] * sa[j];
    }
  }
  float eps = expf(epsP[0]) + 0.03f;
  float gam = expf(gamP[0]);
  float power = gam / (gam + eps);
  bvec[t] = powf((1.0f / MM) / (acc + 1e-8f), power);
}

// a = (prob1 / (K b + 1e-8))^power — block per row, LDS reduction
__global__ void sinkhorn_row(const float* __restrict__ Kmat, const float* __restrict__ bvec,
                             float* __restrict__ avec,
                             const float* __restrict__ epsP, const float* __restrict__ gamP) {
  __shared__ float red[256];
  int bn = blockIdx.x;                  // 0 .. BB*NN-1
  int b = bn / NN;
  int t = threadIdx.x;
  const float* krow = Kmat + (size_t)bn * MM;
  const float* bv = bvec + b * MM;
  float acc = 0.f;
  for (int m = t; m < MM; m += 256) acc += krow[m] * bv[m];
  red[t] = acc; __syncthreads();
  for (int s = 128; s > 0; s >>= 1) {
    if (t < s) red[t] += red[t + s];
    __syncthreads();
  }
  if (t == 0) {
    float eps = expf(epsP[0]) + 0.03f;
    float gam = expf(gamP[0]);
    float power = gam / (gam + eps);
    avec[bn] = powf((1.0f / NN) / (red[0] + 1e-8f), power);
  }
}

// -------------------------- top-32 of T, weighted recon, confidence, flow
__global__ void topk_recon(const float* __restrict__ Kmat, const float* __restrict__ avec,
                           const float* __restrict__ bvec,
                           const float* __restrict__ pc1, const float* __restrict__ pc2,
                           const float* __restrict__ f1n, const float* __restrict__ f2n,
                           float* __restrict__ o_flow, float* __restrict__ o_conf,
                           float* __restrict__ o_recon) {
  int t = blockIdx.x * blockDim.x + threadIdx.x;
  if (t >= BB * NN) return;
  int b = t / NN;
  float a = avec[t];
  const float* krow = Kmat + (size_t)t * MM;
  const float* bv = bvec + b * MM;
  float w[KREC]; int id[KREC];
  for (int i = 0; i < KREC; i++) { w[i] = -1.0f; id[i] = 0; }
  for (int m = 0; m < MM; m++) {
    float Tv = a * krow[m] * bv[m];
    if (Tv > w[KREC-1]) {
      int pos = KREC-1;
      while (pos > 0 && w[pos-1] < Tv) {
        w[pos] = w[pos-1]; id[pos] = id[pos-1]; pos--;
      }
      w[pos] = Tv; id[pos] = m;
    }
  }
  float ws = 0.f;
  for (int i = 0; i < KREC; i++) ws += w[i];
  float inv = 1.0f / (ws + 1e-8f);
  float rx = 0.f, ry = 0.f, rz = 0.f, conf = 0.f;
  const float* f1 = f1n + (size_t)t * 128;
  for (int i = 0; i < KREC; i++) {
    float wn = w[i] * inv;
    const float* q = pc2 + (size_t)(b * MM + id[i]) * 3;
    rx += wn * q[0]; ry += wn * q[1]; rz += wn * q[2];
    const float* f2 = f2n + (size_t)(b * MM + id[i]) * 128;
    float s = 0.f;
    for (int c = 0; c < 128; c++) s += f1[c] * f2[c];   // sim recompute (32 cols only)
    conf += s * wn;
  }
  conf = fmaxf(conf, 0.0f);
  const float* p1 = pc1 + (size_t)t * 3;
  o_flow[t*3+0] = rx - p1[0];
  o_flow[t*3+1] = ry - p1[1];
  o_flow[t*3+2] = rz - p1[2];
  o_conf[t] = conf;
  o_recon[t*3+0] = rx; o_recon[t*3+1] = ry; o_recon[t*3+2] = rz;
}

__global__ void write_idx(const int* __restrict__ idx, float* __restrict__ o, int n) {
  int t = blockIdx.x * blockDim.x + threadIdx.x;
  if (t < n) o[t] = (float)idx[t];
}

// =========================================================== host launcher
struct ConvP {
  const float *b1,*b2,*b3,*be1,*be2,*be3,*g1,*g2,*g3,*w1,*w2,*w3;
};

extern "C" void kernel_launch(void* const* d_in, const int* in_sizes, int n_in,
                              void* d_out, int out_size, void* d_ws, size_t ws_size,
                              hipStream_t stream) {
  (void)in_sizes; (void)n_in; (void)out_size; (void)ws_size;
  const float* pc1 = (const float*)d_in[0];
  const float* pc2 = (const float*)d_in[1];
  // params tree-flattened: conv1..conv3 each with sorted keys
  // b1,b2,b3,be1,be2,be3,g1,g2,g3,w1,w2,w3 ; then epsilon, gamma.
  ConvP conv[3];
  int base = 2;
  for (int c = 0; c < 3; c++) {
    conv[c].b1  = (const float*)d_in[base+0];
    conv[c].b2  = (const float*)d_in[base+1];
    conv[c].b3  = (const float*)d_in[base+2];
    conv[c].be1 = (const float*)d_in[base+3];
    conv[c].be2 = (const float*)d_in[base+4];
    conv[c].be3 = (const float*)d_in[base+5];
    conv[c].g1  = (const float*)d_in[base+6];
    conv[c].g2  = (const float*)d_in[base+7];
    conv[c].g3  = (const float*)d_in[base+8];
    conv[c].w1  = (const float*)d_in[base+9];
    conv[c].w2  = (const float*)d_in[base+10];
    conv[c].w3  = (const float*)d_in[base+11];
    base += 12;
  }
  const float* epsP = (const float*)d_in[base + 0];
  const float* gamP = (const float*)d_in[base + 1];

  // ---- workspace carve-up (aligned 256B)
  size_t off = 0;
  auto carve = [&](size_t bytes) {
    void* p = (char*)d_ws + off;
    off += (bytes + 255) & ~(size_t)255;
    return p;
  };
  int*   idx1   = (int*)  carve((size_t)BB*NN*KNB*4);
  int*   idx2   = (int*)  carve((size_t)BB*MM*KNB*4);
  float* bufX   = (float*)carve((size_t)ROWS_TOTAL*128*4);   // 128 MB
  float* bufY   = (float*)carve((size_t)ROWS_TOTAL*128*4);   // 128 MB
  float* sigbuf = (float*)carve((size_t)BB*NN*128*4);
  float* f1n    = (float*)carve((size_t)BB*NN*128*4);
  float* f2n    = (float*)carve((size_t)BB*MM*128*4);
  _Float16* f1h = (_Float16*)carve((size_t)BB*NN*128*2);
  _Float16* f2h = (_Float16*)carve((size_t)BB*MM*128*2);
  _Float16* Wt  = (_Float16*)carve((size_t)128*128*2);       // packed weights
  float* avec   = (float*)carve((size_t)BB*NN*4);
  float* bvec   = (float*)carve((size_t)BB*MM*4);
  float* stats  = (float*)carve((size_t)BB*128*2*4);
  float* Kmat   = bufX;   // conv scratch is free by the time Sinkhorn runs

  auto statsnorm = [&](float* buf, int C, const float* g, const float* be) {
    zero_stats<<<(BB*C*2 + 255)/256, 256, 0, stream>>>(stats, BB*C*2);
    colstats<<<BB*(ROWS_PER_B/1024), 256, 0, stream>>>(buf, stats, C);
    size_t total = (size_t)ROWS_TOTAL * C;
    norm_act<<<(unsigned)((total + 255)/256), 256, 0, stream>>>(buf, stats, g, be, C);
  };
  auto gemm = [&](const float* A, const float* W, const float* bias, float* Y,
                  int Cin, int CinPad, int Cout) {
    wt_pack<<<(Cout*CinPad + 255)/256, 256, 0, stream>>>(W, Wt, Cin, CinPad, Cout);
    int tiles = (ROWS_TOTAL/16) * (Cout/16);
    gemm_wmma<<<(tiles*32 + 255)/256, 256, 0, stream>>>(A, Wt, bias, Y, CinPad, Cout);
  };

  auto run_cloud = [&](const float* pc, int* idx, float* fn, _Float16* fh) {
    knn_kernel<<<(BB*NN)/256, 256, 0, stream>>>(pc, idx);
    const float* sig = pc;
    int sigC = 3;
    for (int l = 0; l < 3; l++) {
      int Cout = 32 << l;
      int Cin1 = sigC + 3;
      int CinPad1 = (Cin1 + 31) & ~31;     // 6->32, 35->64, 67->96
      const ConvP& P = conv[l];
      // layer 1: materialize gathered+edge input (bufX is free here)
      {
        size_t tot = (size_t)ROWS_TOTAL * CinPad1;
        gather_concat<<<(unsigned)((tot + 255)/256), 256, 0, stream>>>(
            sig, idx, pc, bufX, sigC, CinPad1);
      }
      gemm(bufX, P.w1, P.b1, bufY, Cin1, CinPad1, Cout);
      statsnorm(bufY, Cout, P.g1, P.be1);
      // layer 2
      gemm(bufY, P.w2, P.b2, bufX, Cout, Cout, Cout);
      statsnorm(bufX, Cout, P.g2, P.be2);
      // layer 3
      gemm(bufX, P.w3, P.b3, bufY, Cout, Cout, Cout);
      statsnorm(bufY, Cout, P.g3, P.be3);
      // max over neighbors -> next signal
      size_t tot = (size_t)BB * NN * Cout;
      kmax_kernel<<<(unsigned)((tot + 255)/256), 256, 0, stream>>>(bufY, sigbuf, Cout);
      sig = sigbuf; sigC = Cout;
    }
    l2norm_pack<<<BB*NN, 128, 0, stream>>>(sigbuf, fn, fh);
  };

  run_cloud(pc1, idx1, f1n, f1h);
  run_cloud(pc2, idx2, f2n, f2h);

  // similarity GEMM -> K matrix (L2-resident, 128 MB)
  {
    int waves = BB * (NN/16) * (MM/16);
    sim_k_wmma<<<(waves*32 + 255)/256, 256, 0, stream>>>(f1h, f2h, pc1, pc2, epsP, Kmat);
  }

  // Sinkhorn iterations out of L2
  init_ab<<<(BB*MM + 255)/256, 256, 0, stream>>>(avec, bvec);
  for (int it = 0; it < NB_ITER; it++) {
    sinkhorn_col<<<(BB*MM)/256, 256, 0, stream>>>(Kmat, avec, bvec, epsP, gamP);
    sinkhorn_row<<<BB*NN, 256, 0, stream>>>(Kmat, bvec, avec, epsP, gamP);
  }

  // outputs: recon_flow (B,N,3) | corr_conf (B,N) | target_cross_recon (B,N,3)
  //          | graph_nn_idx (B,N,32)
  float* out = (float*)d_out;
  float* o_flow  = out;
  float* o_conf  = out + (size_t)BB*NN*3;
  float* o_recon = out + (size_t)BB*NN*4;
  float* o_gidx  = out + (size_t)BB*NN*7;
  topk_recon<<<(BB*NN + 63)/64, 64, 0, stream>>>(Kmat, avec, bvec, pc1, pc2,
                                                 f1n, f2n, o_flow, o_conf, o_recon);
  write_idx<<<(BB*NN*KNB + 255)/256, 256, 0, stream>>>(idx1, o_gidx, BB*NN*KNB);
}
